// BandSplitModule_58222576665146
// MI455X (gfx1250) — matrix operand order repaired and checked by reading, not compiled, not run on hardware
//
#include <hip/hip_runtime.h>
#include <hip/hip_bf16.h>

#define NB    37
#define BB    4
#define CC    2
#define FF    1025
#define TT    2048
#define EE    128
#define DD    260      // real feature count = C*MAX_BW*2
#define DPAD  288      // padded to 9 K-chunks of 32
#define NKC   9
#define NPAIR 144      // DPAD/2 packed f16 pairs
#define MAXBW 65
#define EPSLN 1e-5f

#define XS_STRIDE 266  // floats per xs row (bank-conflict pad)
#define YS_STRIDE 296  // halves per ys row (592B = 37*16 -> 16B-aligned rows)

// Per-band W block in the swizzled layout: [kc][ntile][laneB][v]
#define WBAND (NKC * 8 * 32 * 8)   // 18432 dwords

typedef _Float16 v16h __attribute__((ext_vector_type(16)));
typedef float    v8f  __attribute__((ext_vector_type(8)));

union FragU { unsigned u[8]; uint4 q[2]; v16h v; };
union PackU { _Float16 h[2]; unsigned u; };

// ---------------------------------------------------------------------------
// Prep: W[e,d] = g * v / ||v||, f16, swizzled so that a WMMA B-fragment is two
// contiguous 16B chunks per lane:
//   wp[n][kc][nt][laneB][v] = pack_f16( W[e,d0], W[e,d0+1] )
//   with e = nt*16 + (laneB&15), kk = (laneB>=16?8:0)+v, d0 = kc*32 + 2*kk.
// ---------------------------------------------------------------------------
__global__ __launch_bounds__(128) void bs_prep_w(const float* __restrict__ v,
                                                 const float* __restrict__ g,
                                                 unsigned* __restrict__ wp) {
    const int wid  = threadIdx.x >> 5;
    const int lane = threadIdx.x & 31;
    const int row  = blockIdx.x * 4 + wid;          // row = n*128 + e
    if (row >= NB * EE) return;
    const int n = row >> 7, e = row & 127;

    const float* vr = v + (size_t)row * DD;
    float s = 0.f;
    #pragma unroll
    for (int i = 0; i < 9; ++i) {
        int d = lane + 32 * i;
        float t = (d < DD) ? vr[d] : 0.f;
        s += t * t;
    }
    #pragma unroll
    for (int off = 16; off; off >>= 1) s += __shfl_xor(s, off, 32);
    const float scale = (s > 0.f) ? g[row] / sqrtf(s) : 0.f;

    const int nt    = e >> 4;
    const int eLow  = e & 15;
    unsigned* wpn = wp + (size_t)n * WBAND;
    for (int p = lane; p < NPAIR; p += 32) {        // p = kc*16 + kk
        const int kc = p >> 4;
        const int kk = p & 15;
        const int d0 = 2 * p;
        PackU pk;
        pk.h[0] = (d0     < DD) ? (_Float16)(vr[d0]     * scale) : (_Float16)0.f;
        pk.h[1] = (d0 + 1 < DD) ? (_Float16)(vr[d0 + 1] * scale) : (_Float16)0.f;
        const int laneB = ((kk >= 8) ? 16 : 0) + eLow;
        const int vv    = kk & 7;
        wpn[((((size_t)kc * 8 + nt) * 32) + laneB) * 8 + vv] = pk.u;
    }
}

// ---------------------------------------------------------------------------
// Fused gather + masked LayerNorm + f16 WMMA projection.
// Block: 256 threads (8 wave32), one (b, band, 32-time-row) tile.
// ---------------------------------------------------------------------------
__global__ __launch_bounds__(256) void bs_main(const float* __restrict__ x,
                                               const float* __restrict__ gamma,
                                               const float* __restrict__ beta,
                                               const unsigned* __restrict__ wp,
                                               const float* __restrict__ bias,
                                               const int* __restrict__ band_start,
                                               const int* __restrict__ band_width,
                                               float* __restrict__ out) {
    __shared__ __align__(16) float    xs[32 * XS_STRIDE];
    __shared__ __align__(16) _Float16 ys[32 * YS_STRIDE];

    const int tid  = threadIdx.x;
    const int lane = tid & 31;
    const int wid  = tid >> 5;                 // 0..7
    const int n    = blockIdx.y;
    const int b    = blockIdx.z;
    const int t0   = blockIdx.x * 32;
    const int start = band_start[n];
    const int bw    = band_width[n];

    // Pull this band's swizzled W toward L2 early.
    __builtin_prefetch(wp + (size_t)n * WBAND, 0, 1);

    // ---- stage x tile: 32 t-rows x 260 features, coalesced float4 loads ----
    const int grp = tid >> 4;                  // feature-pair subgroup 0..15
    const int sub = tid & 15;                  // time subgroup (2 t's each)
    #pragma unroll
    for (int it = 0; it < 9; ++it) {
        const int pair = it * 16 + grp;        // pair = c*65 + j
        if (pair < CC * MAXBW) {
            const int c = pair >= MAXBW;
            const int j = pair - c * MAXBW;
            int f = start + j; if (f > FF - 1) f = FF - 1;
            const float* src =
                x + ((((size_t)b * CC + c) * FF + f) * TT + (t0 + sub * 2)) * 2;
            const float4 val = *(const float4*)src;   // (t,re)(t,im)(t+1,re)(t+1,im)
            const int tl = sub * 2;
            const int d  = pair * 2;
            *(float2*)&xs[(size_t)tl       * XS_STRIDE + d] = make_float2(val.x, val.y);
            *(float2*)&xs[(size_t)(tl + 1) * XS_STRIDE + d] = make_float2(val.z, val.w);
        }
    }
    __syncthreads();

    // ---- masked LayerNorm: each wave normalizes 4 rows; lane owns 10
    //      contiguous features -> packed dword stores into ys ----
    const float nvalid = (float)(bw * 4);
    const int d0l = lane * 10;
    float gv[10], bv[10], mk[10];
    #pragma unroll
    for (int k = 0; k < 10; ++k) {
        const int d  = d0l + k;
        const bool ok = d < DD;
        gv[k] = ok ? gamma[n * DD + d] : 0.f;
        bv[k] = ok ? beta [n * DD + d] : 0.f;
        const int j = (d % (MAXBW * 2)) >> 1;
        mk[k] = (ok && j < bw) ? 1.f : 0.f;
    }
    #pragma unroll
    for (int rr = 0; rr < 4; ++rr) {
        const int row = wid * 4 + rr;
        const float* xr = &xs[(size_t)row * XS_STRIDE];
        float xv[10], s = 0.f, sq = 0.f;
        #pragma unroll
        for (int k = 0; k < 10; ++k) {
            const int d = d0l + k;
            float val = (d < DD) ? xr[d] : 0.f;
            val *= mk[k];
            xv[k] = val;
            s += val; sq += val * val;
        }
        #pragma unroll
        for (int off = 16; off; off >>= 1) {
            s  += __shfl_xor(s,  off, 32);
            sq += __shfl_xor(sq, off, 32);
        }
        const float mean = s / nvalid;
        const float var  = sq / nvalid - mean * mean;
        const float rs   = rsqrtf(var + EPSLN);
        unsigned* yu = (unsigned*)&ys[(size_t)row * YS_STRIDE];
        #pragma unroll
        for (int m = 0; m < 5; ++m) {
            const int p = lane * 5 + m;        // packs d = d0l+2m, d0l+2m+1
            if (p < NPAIR) {
                PackU pk;
                pk.h[0] = (_Float16)(((xv[2*m]   - mean * mk[2*m]  ) * rs) * gv[2*m]   + bv[2*m]);
                pk.h[1] = (_Float16)(((xv[2*m+1] - mean * mk[2*m+1]) * rs) * gv[2*m+1] + bv[2*m+1]);
                yu[p] = pk.u;                  // zero for p >= 130 -> K-pad
            }
        }
    }
    __syncthreads();

    // ---- WMMA: each wave: one 16-row M-tile x two 16-col N-tiles of E ----
    const int mtile = wid & 1;
    const int ngrp  = wid >> 1;                         // 0..3 -> N-tiles {2g,2g+1}
    const int nt0   = ngrp * 2;
    const int rowA  = mtile * 16 + (lane & 15);
    const unsigned* yrow = (const unsigned*)&ys[(size_t)rowA * YS_STRIDE];
    const int hiA   = (lane >> 4) * 8;                  // +16 halves for lanes 16..31
    const unsigned* wpn = wp + (size_t)n * WBAND;

    v8f acc0 = {}, acc1 = {};
    #pragma unroll
    for (int kc = 0; kc < NKC; ++kc) {
        FragU a;                                        // contiguous k = 16h + s:
        const unsigned* yk = yrow + kc * 16 + hiA;      //   dwords {0..7}
        a.q[0] = *(const uint4*)(yk);
        a.q[1] = *(const uint4*)(yk + 4);

        FragU b0, b1;                                   // swizzled: 2x16B per lane
        const unsigned* wb = wpn + (((size_t)kc * 8 + nt0) * 32 + lane) * 8;
        b0.q[0] = *(const uint4*)(wb);
        b0.q[1] = *(const uint4*)(wb + 4);
        b1.q[0] = *(const uint4*)(wb + 256);            // next N-tile: +32*8 dwords
        b1.q[1] = *(const uint4*)(wb + 260);

        acc0 = __builtin_amdgcn_wmma_f32_16x16x32_f16(false, a.v, false, b0.v,
                                                      (short)0, acc0, false, false);
        acc1 = __builtin_amdgcn_wmma_f32_16x16x32_f16(false, a.v, false, b1.v,
                                                      (short)0, acc1, false, false);
    }

    // ---- epilogue: bias + store, per C/D VGPR layout (M = r + 8*(lane>=16)) ----
    const int eB = nt0 * 16 + (lane & 15);
    const float bias0 = bias[n * EE + eB];
    const float bias1 = bias[n * EE + eB + 16];
    float* outb = out + (((size_t)b * NB + n) * TT) * EE;
    const int tBase = t0 + mtile * 16 + ((lane >> 4) * 8);
    #pragma unroll
    for (int r = 0; r < 8; ++r) {
        float* orow = outb + (size_t)(tBase + r) * EE;
        orow[eB]      = acc0[r] + bias0;
        orow[eB + 16] = acc1[r] + bias1;
    }
}

// ---------------------------------------------------------------------------
extern "C" void kernel_launch(void* const* d_in, const int* in_sizes, int n_in,
                              void* d_out, int out_size, void* d_ws, size_t ws_size,
                              hipStream_t stream) {
    (void)in_sizes; (void)n_in; (void)out_size; (void)ws_size;
    const float* x      = (const float*)d_in[0];
    const float* gamma  = (const float*)d_in[1];
    const float* beta   = (const float*)d_in[2];
    const float* v      = (const float*)d_in[3];
    const float* g      = (const float*)d_in[4];
    const float* bias   = (const float*)d_in[5];
    const int*   bstart = (const int*)d_in[6];
    const int*   bwidth = (const int*)d_in[7];
    unsigned* wp = (unsigned*)d_ws;   // NB*18432 dwords = 2.73 MB scratch

    bs_prep_w<<<dim3((NB * EE + 3) / 4), 128, 0, stream>>>(v, g, wp);
    bs_main<<<dim3(TT / 32, NB, BB), 256, 0, stream>>>(x, gamma, beta, wp, bias,
                                                       bstart, bwidth, (float*)d_out);
}
